// PhyGRU_27504970564219
// MI455X (gfx1250) — matrix operand order repaired and checked by reading, not compiled, not run on hardware
//
#include <hip/hip_runtime.h>
#include <hip/hip_bf16.h>

// Problem constants (from reference)
#define B_    512
#define T_    512
#define SDIM  128          // STATE_DIM
#define LDIM  384          // LATENT_DIM
#define TOT   512          // STATE+LATENT
#define IDIM  128          // INPUT_DIM
#define KTOT  640          // TOT + IDIM  (GEMM K)
#define KB    20           // 640 / 32 k-steps
#define NT1   64           // 1024/16 N tiles, GEMM1 (z 32 | r 24 | phys 8)
#define NT2   24           // 384/16 N tiles, GEMM2
#define FRAG_DW 256        // one B fragment: 32 lanes x 8 dwords
#define G1_FRAGS (NT1*KB)  // 1280
#define G2_FRAGS (NT2*KB)  // 480
#define WS_FRAGS (G1_FRAGS + G2_FRAGS + 1)   // +1 pad frag for pipeline over-read
#define DT_   1.0f

typedef __attribute__((ext_vector_type(16))) __bf16 v16bf;
typedef __attribute__((ext_vector_type(8)))  __bf16 v8bf;
typedef __attribute__((ext_vector_type(8)))  float  v8f;
typedef __attribute__((ext_vector_type(4)))  float  v4f;   // clang vector: OK for nontemporal builtins

__device__ __forceinline__ float bf2f(unsigned short h) {
  union { unsigned int u; float f; } x; x.u = ((unsigned int)h) << 16; return x.f;
}
__device__ __forceinline__ unsigned short f2bf(float f) {
  union { float f; unsigned int u; } x; x.f = f;
  unsigned int r = x.u + 0x7FFFu + ((x.u >> 16) & 1u);
  return (unsigned short)(r >> 16);
}
__device__ __forceinline__ float sigmoidf(float x) { return 1.0f / (1.0f + __expf(-x)); }

__device__ __forceinline__ v16bf cat8(v8bf lo, v8bf hi) {
  return __builtin_shufflevector(lo, hi, 0,1,2,3,4,5,6,7,8,9,10,11,12,13,14,15);
}

// A fragment (16x32 bf16): lane (h=lane>>4, m=lane&15) holds row m,
// K = kb*32 + h*16 + [0..7] (VGPR0-3) and K = kb*32 + h*16 + 8 + [0..7] (VGPR4-7)
__device__ __forceinline__ v16bf load_a_frag(const unsigned short* row, int kb, int h) {
  const v8bf* p0 = (const v8bf*)(row + kb * 32 + h * 16);
  const v8bf* p1 = (const v8bf*)(row + kb * 32 + h * 16 + 8);
  return cat8(*p0, *p1);
}

// B fragment pre-swizzled in d_ws: dword (lane*8 + d) = {W[k0][col], W[k0+1][col]},
// k0 = kb*32 + (lane>>4)*16 + 2d, col = nb*16 + (lane&15)  -> 32x16 bf16 B layout
__device__ __forceinline__ v16bf load_b_frag(const unsigned int* base, int frag, int lane) {
  const v8bf* p = (const v8bf*)(base + (size_t)frag * FRAG_DW + lane * 8);
  return cat8(p[0], p[1]);
}

// Virtual fused GEMM1 weight, K=640 x N=1024 : [Wz.T | Wr.T | (Ap(+)Au zero-padded)]
__device__ __forceinline__ float w1_elem(int k, int j, const float* Ap, const float* Au,
                                         const float* Wr, const float* Wz) {
  if (j < TOT)        return Wz[j * KTOT + k];
  if (j < TOT + LDIM) return Wr[(j - TOT) * KTOT + k];
  int p = j - (TOT + LDIM);
  if (k < SDIM)       return Ap[p * SDIM + k];
  if (k >= TOT)       return Au[p * IDIM + (k - TOT)];
  return 0.0f;
}

// ---- pack: one dword of one fragment per thread ------------------------------
__global__ void phygru_pack(const float* __restrict__ Ap, const float* __restrict__ Au,
                            const float* __restrict__ Wr, const float* __restrict__ Wd,
                            const float* __restrict__ Wz, unsigned int* __restrict__ wpk) {
  int frag = blockIdx.x;
  int tid  = threadIdx.x;
  int lane = tid >> 3;
  int d    = tid & 7;
  int h = lane >> 4, n = lane & 15;
  if (frag >= G1_FRAGS + G2_FRAGS) {      // pad fragment: zero it
    wpk[(size_t)frag * FRAG_DW + lane * 8 + d] = 0u;
    return;
  }
  bool g2 = frag >= G1_FRAGS;
  int f  = g2 ? (frag - G1_FRAGS) : frag;
  int nb = f / KB, kb = f % KB;
  int k0  = kb * 32 + h * 16 + 2 * d;
  int col = nb * 16 + n;
  float lo, hi;
  if (g2) { lo = Wd[col * KTOT + k0];              hi = Wd[col * KTOT + k0 + 1]; }
  else    { lo = w1_elem(k0, col, Ap, Au, Wr, Wz); hi = w1_elem(k0 + 1, col, Ap, Au, Wr, Wz); }
  wpk[(size_t)frag * FRAG_DW + lane * 8 + d] =
      (unsigned int)f2bf(lo) | ((unsigned int)f2bf(hi) << 16);
}

// ---- persistent recurrent kernel: 16 batch rows per WG, 8 waves --------------
__global__ __launch_bounds__(256, 1)
void phygru_scan(const float* __restrict__ xin, const float* __restrict__ hx,
                 const float* __restrict__ br,  const float* __restrict__ bd,
                 const float* __restrict__ bz,  const unsigned int* __restrict__ wpk,
                 float* __restrict__ out) {
  __shared__ __align__(16) unsigned short aT[16][KTOT];   // A tile: [state | u] bf16
  __shared__ __align__(16) unsigned short latC[16][LDIM]; // plain latent copy
  __shared__ __align__(16) unsigned short cand[16][TOT];  // candidate state

  const int tid  = threadIdx.x;
  const int wave = tid >> 5;
  const int lane = tid & 31;
  const int h  = lane >> 4;
  const int n  = lane & 15;
  const int am = lane & 15;                 // A-fragment row owned by this lane
  const int b0 = blockIdx.x * 16;
  const unsigned int* w2 = wpk + (size_t)G1_FRAGS * FRAG_DW;

  // initial state (hx) -> bf16 LDS
  for (int e = tid; e < 16 * TOT; e += 256) {
    int m = e >> 9, c = e & (TOT - 1);
    unsigned short v = f2bf(hx[(size_t)(b0 + m) * TOT + c]);
    aT[m][c] = v;
    if (c >= SDIM) latC[m][c - SDIM] = v;
  }

  // per-lane bias preload (column = tile_base + n)
  float bias1[8];
#pragma unroll
  for (int i = 0; i < 8; ++i) {
    int c = (wave * 8 + i) * 16 + n;
    bias1[i] = (c < TOT) ? bz[c] : ((c < TOT + LDIM) ? br[c - TOT] : 0.0f);
  }
  float bias2[3];
#pragma unroll
  for (int i = 0; i < 3; ++i) bias2[i] = bd[(wave * 3 + i) * 16 + n];

  // u prefetch: thread owns 8 contiguous floats of the 16x128 u tile
  const int um = tid >> 4;                  // tid*8 / 128
  const int uk = (tid * 8) & (IDIM - 1);
  v4f up0, up1;
  {
    const v4f* p = (const v4f*)&xin[((size_t)(b0 + um) * T_ + 0) * IDIM + uk];
    up0 = __builtin_nontemporal_load(p);
    up1 = __builtin_nontemporal_load(p + 1);
  }

  __syncthreads();

  for (int t = 0; t < T_; ++t) {
    // stage u_t (prefetched last iteration) -> aT[:, 512:640]
    {
      unsigned short* dst = &aT[um][TOT + uk];
      dst[0] = f2bf(up0.x); dst[1] = f2bf(up0.y); dst[2] = f2bf(up0.z); dst[3] = f2bf(up0.w);
      dst[4] = f2bf(up1.x); dst[5] = f2bf(up1.y); dst[6] = f2bf(up1.z); dst[7] = f2bf(up1.w);
    }
    // prefetch u_{t+1}: latency hidden behind both GEMMs
    {
      int tn = (t + 1 < T_) ? t + 1 : t;
      const v4f* p = (const v4f*)&xin[((size_t)(b0 + um) * T_ + tn) * IDIM + uk];
      up0 = __builtin_nontemporal_load(p);
      up1 = __builtin_nontemporal_load(p + 1);
    }
    __syncthreads();

    // ---- GEMM1: [z | r | phys_dot] = cat(16x640) x W1(640x1024) --------------
    // double-buffered B fragments + ping-pong A fragment (kb advances by 2)
    v8f acc1[8] = {};
    {
      v16bf a0 = load_a_frag(&aT[am][0], 0, h);
      v16bf b0[8], b1[8];
#pragma unroll
      for (int i = 0; i < 8; ++i) b0[i] = load_b_frag(wpk, (wave * 8 + i) * KB, lane);
#pragma unroll 1
      for (int kb = 0; kb < KB; kb += 2) {
        v16bf a1 = load_a_frag(&aT[am][0], kb + 1, h);
#pragma unroll
        for (int i = 0; i < 8; ++i) b1[i] = load_b_frag(wpk, (wave * 8 + i) * KB + kb + 1, lane);
#pragma unroll
        for (int i = 0; i < 8; ++i)
          acc1[i] = __builtin_amdgcn_wmma_f32_16x16x32_bf16(
              false, a0, false, b0[i], (short)0, acc1[i], false, false);
        a0 = load_a_frag(&aT[am][0], kb + 2, h);            // tail over-read: harmless
#pragma unroll
        for (int i = 0; i < 8; ++i) b0[i] = load_b_frag(wpk, (wave * 8 + i) * KB + kb + 2, lane);
#pragma unroll
        for (int i = 0; i < 8; ++i)
          acc1[i] = __builtin_amdgcn_wmma_f32_16x16x32_bf16(
              false, a1, false, b1[i], (short)0, acc1[i], false, false);
      }
    }
    __syncthreads();   // all GEMM1 reads of aT done before r overwrites latent slot

    if (wave < 4) {            // z owner: sigmoid, keep z in registers across GEMM2
#pragma unroll
      for (int i = 0; i < 8; ++i)
#pragma unroll
        for (int v = 0; v < 8; ++v)
          acc1[i][v] = sigmoidf(acc1[i][v] + bias1[i]);
    } else if (wave < 7) {     // r owner: aT latent slot <- r * latent
#pragma unroll
      for (int i = 0; i < 8; ++i) {
        int lc = (wave * 8 + i) * 16 - TOT + n;
#pragma unroll
        for (int v = 0; v < 8; ++v) {
          int m = v + 8 * h;
          float r = sigmoidf(acc1[i][v] + bias1[i]);
          aT[m][SDIM + lc] = f2bf(r * bf2f(latC[m][lc]));
        }
      }
    } else {                   // phys owner: cand phys <- phys + dt*phys_dot
#pragma unroll
      for (int i = 0; i < 8; ++i) {
        int pc = (wave * 8 + i) * 16 - (TOT + LDIM) + n;
#pragma unroll
        for (int v = 0; v < 8; ++v) {
          int m = v + 8 * h;
          cand[m][pc] = f2bf(bf2f(aT[m][pc]) + DT_ * acc1[i][v]);
        }
      }
    }
    __syncthreads();

    // ---- GEMM2: latent_dot = [phys, r*latent, u] x Wd.T(640x384) -------------
    v8f acc2[3] = {};
    {
      v16bf a0 = load_a_frag(&aT[am][0], 0, h);
      v16bf b0[3], b1[3];
#pragma unroll
      for (int i = 0; i < 3; ++i) b0[i] = load_b_frag(w2, (wave * 3 + i) * KB, lane);
#pragma unroll 1
      for (int kb = 0; kb < KB; kb += 2) {
        v16bf a1 = load_a_frag(&aT[am][0], kb + 1, h);
#pragma unroll
        for (int i = 0; i < 3; ++i) b1[i] = load_b_frag(w2, (wave * 3 + i) * KB + kb + 1, lane);
#pragma unroll
        for (int i = 0; i < 3; ++i)
          acc2[i] = __builtin_amdgcn_wmma_f32_16x16x32_bf16(
              false, a0, false, b0[i], (short)0, acc2[i], false, false);
        a0 = load_a_frag(&aT[am][0], kb + 2, h);            // tail over-read: harmless
#pragma unroll
        for (int i = 0; i < 3; ++i) b0[i] = load_b_frag(w2, (wave * 3 + i) * KB + kb + 2, lane);
#pragma unroll
        for (int i = 0; i < 3; ++i)
          acc2[i] = __builtin_amdgcn_wmma_f32_16x16x32_bf16(
              false, a1, false, b1[i], (short)0, acc2[i], false, false);
      }
    }
#pragma unroll
    for (int i = 0; i < 3; ++i) {       // cand latent <- latent + dt*(dot + bd)
      int lc = (wave * 3 + i) * 16 + n;
#pragma unroll
      for (int v = 0; v < 8; ++v) {
        int m = v + 8 * h;
        cand[m][SDIM + lc] = f2bf(bf2f(latC[m][lc]) + DT_ * (acc2[i][v] + bias2[i]));
      }
    }
    __syncthreads();

    // ---- gated update + output store (z-owner waves own state columns) -------
    if (wave < 4) {
#pragma unroll
      for (int i = 0; i < 8; ++i) {
        int c = (wave * 8 + i) * 16 + n;
#pragma unroll
        for (int v = 0; v < 8; ++v) {
          int m = v + 8 * h;
          float z = acc1[i][v];
          float oldv = (c < SDIM) ? bf2f(aT[m][c]) : bf2f(latC[m][c - SDIM]);
          float ns = z * bf2f(cand[m][c]) + (1.0f - z) * oldv;
          __builtin_nontemporal_store(ns, &out[((size_t)(b0 + m) * T_ + t) * TOT + c]);
          unsigned short nb16 = f2bf(ns);
          aT[m][c] = nb16;
          if (c >= SDIM) latC[m][c - SDIM] = nb16;
          if (t == T_ - 1)
            out[(size_t)B_ * T_ * TOT + (size_t)(b0 + m) * TOT + c] = ns;
        }
      }
    }
    __syncthreads();
  }
}

extern "C" void kernel_launch(void* const* d_in, const int* in_sizes, int n_in,
                              void* d_out, int out_size, void* d_ws, size_t ws_size,
                              hipStream_t stream) {
  (void)in_sizes; (void)n_in; (void)out_size;
  const float* xin = (const float*)d_in[0];  // inputs (B,T,128)
  const float* hx  = (const float*)d_in[1];  // (B,512)
  const float* Ap  = (const float*)d_in[2];
  const float* Au  = (const float*)d_in[3];
  const float* Wr  = (const float*)d_in[4];
  const float* br  = (const float*)d_in[5];
  const float* Wd  = (const float*)d_in[6];
  const float* bd  = (const float*)d_in[7];
  const float* Wz  = (const float*)d_in[8];
  const float* bz  = (const float*)d_in[9];
  unsigned int* wpk = (unsigned int*)d_ws;
  float* out = (float*)d_out;

  if (ws_size < (size_t)WS_FRAGS * FRAG_DW * sizeof(unsigned int)) return;

  phygru_pack<<<WS_FRAGS, 256, 0, stream>>>(Ap, Au, Wr, Wd, Wz, wpk);
  phygru_scan<<<B_ / 16, 256, 0, stream>>>(xin, hx, br, bd, bz, wpk, out);
}
